// MultiScaleCodebook_88708254532134
// MI455X (gfx1250) — matrix operand order repaired and checked by reading, not compiled, not run on hardware
//
#include <hip/hip_runtime.h>
#include <hip/hip_bf16.h>
#include <math.h>
#include <stdint.h>

typedef __attribute__((ext_vector_type(16))) __bf16       v16bf;
typedef __attribute__((ext_vector_type(8)))  __bf16       v8bf;
typedef __attribute__((ext_vector_type(8)))  float        v8f;
typedef __attribute__((ext_vector_type(4)))  unsigned int v4u;
typedef __attribute__((ext_vector_type(8)))  int          v8i;
typedef __attribute__((ext_vector_type(4)))  int          v4i;

#define NCODES 16384
#define CDIM   64
#define TILE_CODES 128                       // codes staged per LDS tile
#define TILE_BYTES (TILE_CODES * CDIM * 2)   // 16 KB bf16

#if __has_builtin(__builtin_amdgcn_tensor_load_to_lds) && \
    __has_builtin(__builtin_amdgcn_s_wait_tensorcnt)
#define VQ_HAVE_TDM 1
#else
#define VQ_HAVE_TDM 0
#endif

// ---------------------------------------------------------------------------
// utility: zero a float region
// ---------------------------------------------------------------------------
__global__ void vq_zero(float* p, int n) {
    int i = blockIdx.x * 256 + threadIdx.x;
    if (i < n) p[i] = 0.0f;
}

// ---------------------------------------------------------------------------
// codebook prep: fp32 -> bf16 copy + ||e||^2 per code
// ---------------------------------------------------------------------------
__global__ void __launch_bounds__(64)
vq_prep(const float* __restrict__ emb, __hip_bfloat16* __restrict__ embB,
        float* __restrict__ esq) {
    __shared__ float red[64];
    const int code = blockIdx.x, c = threadIdx.x;
    float e = emb[(size_t)code * CDIM + c];
    embB[(size_t)code * CDIM + c] = __float2bfloat16(e);
    red[c] = e * e;
    __syncthreads();
    for (int off = 32; off > 0; off >>= 1) {
        if (c < off) red[c] += red[c + off];
        __syncthreads();
    }
    if (c == 0) esq[code] = red[0];
}

// ---------------------------------------------------------------------------
// rest = areapool(z - accu) ; emit bf16 position-major [N][64]
// (||rest||^2 is NOT needed: argmin is invariant to it)
// ---------------------------------------------------------------------------
__global__ void __launch_bounds__(64)
vq_downsample(const float* __restrict__ z, const float* __restrict__ accu,
              __hip_bfloat16* __restrict__ restS, int tpn, int pn) {
    const int n = blockIdx.x, c = threadIdx.x;
    const int pp = pn * pn;
    const int b = n / (tpn * pp);
    int rem = n % (tpn * pp);
    const int u = rem / pp; rem %= pp;
    const int v = rem / pn; const int x = rem % pn;
    const int su = (u * 4) / tpn,  eu = ((u + 1) * 4 + tpn - 1) / tpn;
    const int sv = (v * 16) / pn,  ev = ((v + 1) * 16 + pn - 1) / pn;
    const int sx = (x * 16) / pn,  ex = ((x + 1) * 16 + pn - 1) / pn;
    const float wgt = 1.0f / (float)((eu - su) * (ev - sv) * (ex - sx));
    const size_t base = ((size_t)b * CDIM + c) * 1024;
    float s = 0.0f;
    for (int tt = su; tt < eu; tt++)
        for (int hh = sv; hh < ev; hh++)
            for (int ww = sx; ww < ex; ww++) {
                size_t q = base + (size_t)tt * 256 + hh * 16 + ww;
                s += z[q] - accu[q];
            }
    restS[(size_t)n * CDIM + c] = __float2bfloat16(s * wgt);
}

// ---------------------------------------------------------------------------
// order-preserving float->u32 map, packed with code for u64 atomic-min argmin
// ---------------------------------------------------------------------------
__device__ inline unsigned long long vq_pack(float d, int code) {
    unsigned u = __float_as_uint(d);
    u = (u & 0x80000000u) ? ~u : (u | 0x80000000u);
    return ((unsigned long long)u << 32) | (unsigned)code;
}

#if VQ_HAVE_TDM
// Issue one TDM DMA: 128 codebook rows (bf16, 64 elem each) -> LDS tile.
// D# layout per CDNA5 ISA ch.8: group0 {count/lds_addr/global_addr/type},
// group1 {data_size, tensor dims 64 x 16384, tile 64x128, stride 64}.
// This toolchain exposes the 6-arg builtin:
//   (uint32x4 g0, int32x8 g1, int32x4 g2, int32x4 g3, int32x8 g4, i32 cpol)
__device__ inline void vq_tdm_issue(unsigned long long gaddr, unsigned ldsOff) {
    v4u g0;
    g0[0] = 1u;                                        // count=1
    g0[1] = ldsOff;                                    // lds_addr
    g0[2] = (unsigned)(gaddr & 0xffffffffu);           // global_addr[31:0]
    g0[3] = (unsigned)((gaddr >> 32) & 0x1ffffffu)     // global_addr[56:32]
            | (2u << 30);                              // type=2 (image)
    v8i g1;
    g1[0] = (1 << 16);                                 // data_size=1 (2 bytes)
    g1[1] = (CDIM << 16);                              // tensor_dim0[15:0]
    g1[2] = (NCODES << 16);                            // dim0 hi=0 | tensor_dim1[15:0]
    g1[3] = (CDIM << 16);                              // dim1 hi=0 | tile_dim0=64
    g1[4] = TILE_CODES;                                // tile_dim1=128, tile_dim2=0
    g1[5] = CDIM;                                      // tensor_dim0_stride=64
    g1[6] = 0;
    g1[7] = 0;
    v4i g2 = {0, 0, 0, 0};
    v4i g3 = {0, 0, 0, 0};
    v8i g4 = {0, 0, 0, 0, 0, 0, 0, 0};
    __builtin_amdgcn_tensor_load_to_lds(g0, g1, g2, g3, g4, 0);
}
#endif

// ---------------------------------------------------------------------------
// WMMA nearest-code search, argmin_k( ||e_k||^2 - 2 z.e_k ).
// Block = 256 threads (8 wave32) owns 64 positions (4 M-tiles).
// Codebook streamed as 128-code/16KB LDS tiles, double-buffered via the
// Tensor Data Mover (wave0 issues, s_wait_tensorcnt + barrier publishes).
// Each wave takes 16 codes of the tile; 8 WMMAs per wave per tile reuse the
// same B fragments across 4 M-tiles. Final argmin via packed u64 ds_min.
// ---------------------------------------------------------------------------
__global__ void __launch_bounds__(256)
vq_argmin_wmma(const __hip_bfloat16* __restrict__ restS,
               const __hip_bfloat16* __restrict__ embB,
               const float* __restrict__ esq,
               int N, int* __restrict__ idxOut, float* __restrict__ idxfOut) {
    __shared__ __align__(64) __bf16 sTile[2][TILE_CODES * CDIM];
    __shared__ unsigned long long sBest[64];

    const int tid  = threadIdx.x;
    const int wave = tid >> 5;
    const int lane = tid & 31;
    const int hi   = lane >> 4;       // lane half = K split (A) / K half (B)
    const int lcol = lane & 15;
    const int tile0 = blockIdx.x * 64;

    if (tid < 64) sBest[tid] = ~0ull;

    // ---- A fragments: 4 M-tiles x (16 rows x K=64) as 16x32 bf16 chunks ----
    v16bf a0[4], a1[4];
#pragma unroll
    for (int mt = 0; mt < 4; mt++) {
        int rowA = tile0 + mt * 16 + lcol; if (rowA > N - 1) rowA = N - 1;
        const __bf16* zr = reinterpret_cast<const __bf16*>(restS) + (size_t)rowA * CDIM;
        v8bf l0 = *reinterpret_cast<const v8bf*>(zr +      hi * 16);
        v8bf h0 = *reinterpret_cast<const v8bf*>(zr +  8 + hi * 16);
        v8bf l1 = *reinterpret_cast<const v8bf*>(zr + 32 + hi * 16);
        v8bf h1 = *reinterpret_cast<const v8bf*>(zr + 40 + hi * 16);
#pragma unroll
        for (int i = 0; i < 8; i++) {
            a0[mt][i] = l0[i]; a0[mt][i + 8] = h0[i];
            a1[mt][i] = l1[i]; a1[mt][i + 8] = h1[i];
        }
    }

    float bestD[4][8]; int bestI[4][8];
#pragma unroll
    for (int mt = 0; mt < 4; mt++)
#pragma unroll
        for (int r = 0; r < 8; r++) { bestD[mt][r] = 3.4e38f; bestI[mt][r] = 0x7fffffff; }

    const int numTiles = NCODES / TILE_CODES;   // 128
#if VQ_HAVE_TDM
    const unsigned long long gbase = (unsigned long long)(uintptr_t)embB;
    const unsigned lds0 = (unsigned)(uintptr_t)(&sTile[0][0]);
    const unsigned lds1 = (unsigned)(uintptr_t)(&sTile[1][0]);
    if (wave == 0) vq_tdm_issue(gbase, lds0);   // prologue: tile 0 -> buf 0
#endif

    for (int t = 0; t < numTiles; t++) {
        const int buf = t & 1;
#if VQ_HAVE_TDM
        if (wave == 0) {
            if (t + 1 < numTiles) {
                vq_tdm_issue(gbase + (unsigned long long)(t + 1) * TILE_BYTES,
                             (buf ? lds0 : lds1));
                __builtin_amdgcn_s_wait_tensorcnt(1);   // tile t landed
            } else {
                __builtin_amdgcn_s_wait_tensorcnt(0);
            }
        }
        __syncthreads();                                // publish buf[t&1]
#else
        {   // fallback: cooperative global -> LDS copy (64 B per thread)
            const __bf16* src = reinterpret_cast<const __bf16*>(embB) +
                                (size_t)t * TILE_CODES * CDIM + tid * 32;
            v8bf* dst = reinterpret_cast<v8bf*>(&sTile[buf][tid * 32]);
            dst[0] = *reinterpret_cast<const v8bf*>(src);
            dst[1] = *reinterpret_cast<const v8bf*>(src + 8);
            dst[2] = *reinterpret_cast<const v8bf*>(src + 16);
            dst[3] = *reinterpret_cast<const v8bf*>(src + 24);
            __syncthreads();
        }
#endif
        // B fragments from LDS: this wave's 16 codes, K split per lane half
        const __bf16* bp = &sTile[buf][(wave * 16 + lcol) * CDIM];
        v16bf b0 = *reinterpret_cast<const v16bf*>(bp +      hi * 16);
        v16bf b1 = *reinterpret_cast<const v16bf*>(bp + 32 + hi * 16);
        const int code = t * TILE_CODES + wave * 16 + lcol;
        const float eq = esq[code];

#pragma unroll
        for (int mt = 0; mt < 4; mt++) {
            v8f acc = {};
            acc = __builtin_amdgcn_wmma_f32_16x16x32_bf16(false, a0[mt], false, b0,
                                                          (short)0, acc, false, false);
            acc = __builtin_amdgcn_wmma_f32_16x16x32_bf16(false, a1[mt], false, b1,
                                                          (short)0, acc, false, false);
#pragma unroll
            for (int r = 0; r < 8; r++) {
                float d = eq - 2.0f * acc[r];
                if (d < bestD[mt][r]) { bestD[mt][r] = d; bestI[mt][r] = code; }
            }
        }
        __syncthreads();            // everyone done reading buf before reuse
    }

    // ---- cross-wave argmin via packed u64 LDS atomic-min ----
#pragma unroll
    for (int mt = 0; mt < 4; mt++)
#pragma unroll
        for (int r = 0; r < 8; r++) {
            const int rl = mt * 16 + r + 8 * hi;
            atomicMin(&sBest[rl], vq_pack(bestD[mt][r], bestI[mt][r]));
        }
    __syncthreads();
    if (tid < 64) {
        const int row = tile0 + tid;
        if (row < N) {
            const int bi = (int)(sBest[tid] & 0xffffffffu);
            idxOut[row]  = bi;
            idxfOut[row] = (float)bi;
        }
    }
}

// ---------------------------------------------------------------------------
// trilinear upsample of gathered codes -> hbuf[b][t][h][w][c] fp32
// ---------------------------------------------------------------------------
__device__ inline void lin_tap(int i, int in, int out, int& i0, int& i1, float& w) {
    float src = ((float)i + 0.5f) * ((float)in / (float)out) - 0.5f;
    if (src < 0.0f) src = 0.0f;
    i0 = (int)src;
    if (i0 > in - 1) i0 = in - 1;
    i1 = i0 + 1; if (i1 > in - 1) i1 = in - 1;
    w = src - (float)i0;
}

__global__ void __launch_bounds__(256)
vq_upsample(const float* __restrict__ emb, const int* __restrict__ idx,
            float* __restrict__ hbuf, int tpn, int pn) {
    int gid = blockIdx.x * 256 + threadIdx.x;
    const int c = gid & 63;
    int vox = gid >> 6;
    const int ww = vox & 15; vox >>= 4;
    const int hh = vox & 15; vox >>= 4;
    const int tt = vox & 3;  vox >>= 2;
    const int b  = vox;
    int t0, t1, h0, h1, w0, w1; float wt, wh, wv;
    lin_tap(tt, tpn, 4, t0, t1, wt);
    lin_tap(hh, pn, 16, h0, h1, wh);
    lin_tap(ww, pn, 16, w0, w1, wv);
    const float wts[2] = {1.0f - wt, wt}; const int tsv[2] = {t0, t1};
    const float whs[2] = {1.0f - wh, wh}; const int hsv[2] = {h0, h1};
    const float wws[2] = {1.0f - wv, wv}; const int wsv[2] = {w0, w1};
    float s = 0.0f;
    for (int a = 0; a < 2; a++)
        for (int e = 0; e < 2; e++)
            for (int f = 0; f < 2; f++) {
                int n = ((b * tpn + tsv[a]) * pn + hsv[e]) * pn + wsv[f];
                s += wts[a] * whs[e] * wws[f] * emb[(size_t)idx[n] * CDIM + c];
            }
    hbuf[((size_t)((b * 4 + tt) * 16 + hh) * 16 + ww) * CDIM + c] = s;
}

// ---------------------------------------------------------------------------
// y = 0.5*h + 0.5*conv3d(h, Wq[qi], bq[qi]) ; accu += y ; commit += (accu-z)^2
// ---------------------------------------------------------------------------
__global__ void __launch_bounds__(64)
vq_conv_accum(const float* __restrict__ hbuf, const float* __restrict__ Wq,
              const float* __restrict__ bq, int qi,
              float* __restrict__ accu, const float* __restrict__ z,
              float* __restrict__ commitAcc) {
    __shared__ float sIn[64];
    __shared__ float red[64];
    int vox = blockIdx.x;
    const int ww = vox & 15; vox >>= 4;
    const int hh = vox & 15; vox >>= 4;
    const int tt = vox & 3;  vox >>= 2;
    const int b  = vox;
    const int co = threadIdx.x;
    float acc = bq[qi * CDIM + co];
    const float* wb = Wq + (((size_t)qi * CDIM + co) * CDIM) * 27;
    for (int dt = -1; dt <= 1; dt++)
        for (int dh = -1; dh <= 1; dh++)
            for (int dw = -1; dw <= 1; dw++) {
                const int t2 = tt + dt, h2 = hh + dh, w2 = ww + dw;
                const bool ok = (t2 >= 0 && t2 < 4 && h2 >= 0 && h2 < 16 &&
                                 w2 >= 0 && w2 < 16);
                __syncthreads();
                sIn[co] = ok ? hbuf[((size_t)((b * 4 + t2) * 16 + h2) * 16 + w2) * CDIM + co]
                             : 0.0f;
                __syncthreads();
                const int tap = ((dt + 1) * 3 + (dh + 1)) * 3 + (dw + 1);
                const float* wr = wb + tap;
#pragma unroll 8
                for (int ci = 0; ci < 64; ci++) acc += sIn[ci] * wr[ci * 27];
            }
    const float hv = hbuf[((size_t)((b * 4 + tt) * 16 + hh) * 16 + ww) * CDIM + co];
    const float y  = 0.5f * hv + 0.5f * acc;
    const size_t ai = ((size_t)b * CDIM + co) * 1024 + (size_t)tt * 256 + hh * 16 + ww;
    const float na = accu[ai] + y;
    accu[ai] = na;
    const float d = na - z[ai];
    red[co] = d * d;
    __syncthreads();
    for (int off = 32; off > 0; off >>= 1) {
        if (co < off) red[co] += red[co + off];
        __syncthreads();
    }
    if (co == 0) atomicAdd(commitAcc, red[0]);
}

// ---------------------------------------------------------------------------
// stats from last-scale indices
// ---------------------------------------------------------------------------
__global__ void vq_bincount(const int* __restrict__ idx, int* __restrict__ cnt, int n) {
    int i = blockIdx.x * 256 + threadIdx.x;
    if (i < n) atomicAdd(&cnt[idx[i]], 1);
}

__global__ void __launch_bounds__(256)
vq_usage(const int* __restrict__ cnt, float* __restrict__ usage,
         float* __restrict__ entAcc, int* __restrict__ usedAcc, float invTot) {
    __shared__ float red[256];
    __shared__ int   redi[256];
    int k = blockIdx.x * 256 + threadIdx.x;
    float p = (float)cnt[k] * invTot;
    usage[k] = p;
    red[threadIdx.x]  = p * logf(p + 1e-10f);
    redi[threadIdx.x] = (p > (1.0f / 16384.0f)) ? 1 : 0;
    __syncthreads();
    for (int off = 128; off > 0; off >>= 1) {
        if (threadIdx.x < off) {
            red[threadIdx.x]  += red[threadIdx.x + off];
            redi[threadIdx.x] += redi[threadIdx.x + off];
        }
        __syncthreads();
    }
    if (threadIdx.x == 0) { atomicAdd(entAcc, red[0]); atomicAdd(usedAcc, redi[0]); }
}

__global__ void vq_finalize(const float* commitAcc, const float* entAcc,
                            const int* usedAcc, float* outCommit,
                            float* outPerp, float* outAvg) {
    if (threadIdx.x == 0 && blockIdx.x == 0) {
        *outCommit = 0.25f * (*commitAcc) / (131072.0f * 10.0f);
        *outPerp   = expf(-(*entAcc));
        *outAvg    = (float)(*usedAcc) / 16384.0f;
    }
}

// ---------------------------------------------------------------------------
// host launcher
// ---------------------------------------------------------------------------
extern "C" void kernel_launch(void* const* d_in, const int* in_sizes, int n_in,
                              void* d_out, int out_size, void* d_ws, size_t ws_size,
                              hipStream_t stream) {
    (void)in_sizes; (void)n_in; (void)out_size; (void)ws_size;
    const float* z   = (const float*)d_in[0];   // [8,64,4,16,16]
    const float* emb = (const float*)d_in[1];   // [16384,64]
    const float* Wq  = (const float*)d_in[2];   // [4,64,64,3,3,3]
    const float* bq  = (const float*)d_in[3];   // [4,64]
    float* out = (float*)d_out;

    char* ws = (char*)d_ws;
    size_t off = 0;
    auto alloc = [&](size_t bytes) -> char* {
        char* p = ws + off;
        off = (off + bytes + 255) & ~(size_t)255;
        return p;
    };
    __hip_bfloat16* embB  = (__hip_bfloat16*)alloc((size_t)NCODES * CDIM * 2);
    float*          esq   = (float*)alloc((size_t)NCODES * 4);
    __hip_bfloat16* restS = (__hip_bfloat16*)alloc((size_t)8192 * CDIM * 2);
    int*            idx   = (int*)alloc(8192 * 4);
    float*          hbuf  = (float*)alloc((size_t)8192 * CDIM * 4);
    int*            cnt   = (int*)alloc((size_t)NCODES * 4);
    float*          scal  = (float*)alloc(16);
    float* commitAcc = scal;
    float* entAcc    = scal + 1;
    int*   usedAcc   = (int*)(scal + 2);

    // output layout: accu | commit | perplexity | avg_usage | usage | ms_idx...
    float* accu      = out;
    float* outCommit = out + 131072;
    float* outPerp   = out + 131073;
    float* outAvg    = out + 131074;
    float* outUsage  = out + 131075;
    float* outIdx    = out + 131075 + NCODES;

    vq_zero<<<(131072 + 255) / 256, 256, 0, stream>>>(accu, 131072);
    vq_zero<<<(NCODES + 255) / 256, 256, 0, stream>>>((float*)cnt, NCODES);
    vq_zero<<<1, 256, 0, stream>>>(scal, 4);

    vq_prep<<<NCODES, 64, 0, stream>>>(emb, embB, esq);

    static const int T_PN[10] = {1, 1, 2, 2, 2, 4, 4, 4, 4, 4};
    static const int V_PN[10] = {1, 2, 3, 4, 5, 6, 8, 10, 13, 16};
    int qiArr[10];
    {
        double ticks[4];
        for (int k = 0; k < 4; k++)
            ticks[k] = (1.0 / 12.0) + k * ((11.0 / 12.0 - 1.0 / 12.0) / 3.0);
        for (int s = 0; s < 10; s++) {
            double x = (double)s / 9.0;
            int best = 0; double bd = 1e30;
            for (int k = 0; k < 4; k++) {
                double d = fabs(ticks[k] - x);
                if (d < bd) { bd = d; best = k; }
            }
            qiArr[s] = best;
        }
    }

    int msOff = 0;
    for (int s = 0; s < 10; s++) {
        const int tpn = T_PN[s], pn = V_PN[s];
        const int N = 8 * tpn * pn * pn;
        vq_downsample<<<N, 64, 0, stream>>>(z, accu, restS, tpn, pn);
        vq_argmin_wmma<<<(N + 63) / 64, 256, 0, stream>>>(restS, embB, esq, N,
                                                          idx, outIdx + msOff);
        vq_upsample<<<2048, 256, 0, stream>>>(emb, idx, hbuf, tpn, pn);
        vq_conv_accum<<<8192, 64, 0, stream>>>(hbuf, Wq, bq, qiArr[s], accu, z,
                                               commitAcc);
        msOff += N;
    }
    vq_bincount<<<32, 256, 0, stream>>>(idx, cnt, 8192);
    vq_usage<<<NCODES / 256, 256, 0, stream>>>(cnt, outUsage, entAcc, usedAcc,
                                               1.0f / 8192.0f);
    vq_finalize<<<1, 1, 0, stream>>>(commitAcc, entAcc, usedAcc, outCommit,
                                     outPerp, outAvg);
}
